// DQN_86449101733992
// MI455X (gfx1250) — matrix operand order repaired and checked by reading, not compiled, not run on hardware
//
#include <hip/hip_runtime.h>
#include <hip/hip_bf16.h>
#include <stdint.h>

// ---------------------------------------------------------------------------
// DRQN forward for MI455X (gfx1250, wave32, WMMA).
// GEMM work: v_wmma_f32_16x16x32_bf16.  LDS staging: TDM (tensor_load_to_lds,
// TENSORcnt) for conv2, async global->LDS (ASYNCcnt) for conv3.
// ---------------------------------------------------------------------------

typedef __bf16 bf16_t;
typedef __attribute__((ext_vector_type(16))) __bf16 v16bf;
typedef __attribute__((ext_vector_type(8)))  __bf16 v8bf;
typedef __attribute__((ext_vector_type(8)))  float  v8f;
typedef __attribute__((ext_vector_type(4)))  uint32_t u32x4;
typedef __attribute__((ext_vector_type(8)))  uint32_t u32x8;

#define WMMA_BF16(a, b, c) \
  __builtin_amdgcn_wmma_f32_16x16x32_bf16(false, (a), false, (b), (short)0, (c), false, false)

// A-fragment (16x32 bf16, M x K): lane<16 holds K={0..7,16..23},
// lane>=16 holds K={8..15,24..31} of row M=lane&15.  Row-major [M][Kpad].
__device__ __forceinline__ v16bf load_a_frag(const bf16_t* __restrict__ row, int kbase, int hi) {
  const bf16_t* p = row + kbase + (hi ? 8 : 0);
  v8bf lo = *(const v8bf*)(p);
  v8bf hh = *(const v8bf*)(p + 16);
  v16bf f;
#pragma unroll
  for (int j = 0; j < 8; ++j) { f[j] = lo[j]; f[j + 8] = hh[j]; }
  return f;
}

// B-fragment (32x16 bf16, K x N): lane = column n, same K order as A
// (lanes 0-15: K={0..7,16..23}, lanes 16-31: K={8..15,24..31}).  Weights stored [N][Kpad].
__device__ __forceinline__ v16bf load_b_frag(const bf16_t* __restrict__ row, int kbase, int hi) {
  const bf16_t* p = row + kbase + (hi ? 8 : 0);
  v8bf lo = *(const v8bf*)(p);
  v8bf hh = *(const v8bf*)(p + 16);
  v16bf f;
#pragma unroll
  for (int j = 0; j < 8; ++j) { f[j] = lo[j]; f[j + 8] = hh[j]; }
  return f;
}

// ---------------------------------------------------------------------------
// Register-blocked GEMM: wave computes an MB x NB grid of 16x16 tiles.
// out[M][N] = relu?( A[M][Kpad] * B[N][Kpad]^T + bias[N] )
// ---------------------------------------------------------------------------
template <int MB, int NB>
__global__ void gemm_bf16_wmma(const bf16_t* __restrict__ A, const bf16_t* __restrict__ Bw,
                               const float* __restrict__ bias,
                               float* __restrict__ outF, bf16_t* __restrict__ outB,
                               int Kpad, int ksteps, int ngrp_total, int tiles,
                               int out_stride, int relu) {
  int wave = blockIdx.x * (blockDim.x >> 5) + (threadIdx.x >> 5);
  if (wave >= tiles) return;
  int mg = wave / ngrp_total, ng = wave - mg * ngrp_total;
  int lane = threadIdx.x & 31, l15 = lane & 15, hi = lane >> 4;
  const bf16_t* arow[MB];
  const bf16_t* brow[NB];
#pragma unroll
  for (int mi = 0; mi < MB; ++mi)
    arow[mi] = A + (size_t)((mg * MB + mi) * 16 + l15) * Kpad;
#pragma unroll
  for (int ni = 0; ni < NB; ++ni)
    brow[ni] = Bw + (size_t)((ng * NB + ni) * 16 + l15) * Kpad;

  v8f acc[MB][NB];
#pragma unroll
  for (int mi = 0; mi < MB; ++mi)
#pragma unroll
    for (int ni = 0; ni < NB; ++ni) acc[mi][ni] = (v8f){};

  for (int ks = 0; ks < ksteps; ++ks) {
    int kb = ks * 32;
    if (ks + 4 < ksteps) {
#pragma unroll
      for (int mi = 0; mi < MB; ++mi)
        __builtin_prefetch((const void*)(arow[mi] + kb + 128), 0, 0);
    }
    v16bf av[MB], bv[NB];
#pragma unroll
    for (int mi = 0; mi < MB; ++mi) av[mi] = load_a_frag(arow[mi], kb, hi);
#pragma unroll
    for (int ni = 0; ni < NB; ++ni) bv[ni] = load_b_frag(brow[ni], kb, hi);
#pragma unroll
    for (int mi = 0; mi < MB; ++mi)
#pragma unroll
      for (int ni = 0; ni < NB; ++ni)
        acc[mi][ni] = WMMA_BF16(av[mi], bv[ni], acc[mi][ni]);
  }

#pragma unroll
  for (int ni = 0; ni < NB; ++ni) {
    int n = (ng * NB + ni) * 16 + l15;
    float bz = bias ? bias[n] : 0.f;
#pragma unroll
    for (int mi = 0; mi < MB; ++mi) {
#pragma unroll
      for (int r = 0; r < 8; ++r) {
        int m = (mg * MB + mi) * 16 + (hi ? 8 : 0) + r;
        float v = acc[mi][ni][r] + bz;
        if (relu) v = v > 0.f ? v : 0.f;
        size_t o = (size_t)m * out_stride + n;
        if (outF) outF[o] = v; else outB[o] = (bf16_t)v;
      }
    }
  }
}

// ---------------------------------------------------------------------------
// Conv1: obs stacking fused.  Per sample (t,b): in = obs[t..t+3, b] f32,
// out [32][20][20] bf16.  K = s*64+ky*8+kx (256), implicit-GEMM, stride 4.
// Wave per (sample, m-tile).  A-gather chunks are 8 contiguous f32.
// ---------------------------------------------------------------------------
__global__ void conv1_wmma(const float* __restrict__ obs, const bf16_t* __restrict__ wb1,
                           const float* __restrict__ b1, bf16_t* __restrict__ out) {
  int wave = blockIdx.x * (blockDim.x >> 5) + (threadIdx.x >> 5);
  int sample = wave / 25, mt = wave - sample * 25;
  int t = sample >> 6, b = sample & 63;
  int lane = threadIdx.x & 31, l15 = lane & 15, hi = lane >> 4;
  int m = mt * 16 + l15;
  int py = m / 20, px = m - py * 20;
  const bf16_t* brow0 = wb1 + (size_t)l15 * 256;
  const bf16_t* brow1 = wb1 + (size_t)(16 + l15) * 256;
  v8f acc0 = {}, acc1 = {};
  for (int ks = 0; ks < 8; ++ks) {
    int kb = ks * 32;
    v16bf a;
#pragma unroll
    for (int c = 0; c < 2; ++c) {
      int kc = kb + (hi ? 8 : 0) + c * 16;     // 8 contiguous K: fixed (s,ky), kx=0..7
      int s = kc >> 6, ky = (kc >> 3) & 7;
      const float* src = obs + (size_t)((t + s) * 64 + b) * 7056 + (py * 4 + ky) * 84 + px * 4;
#pragma unroll
      for (int j = 0; j < 8; ++j) a[c * 8 + j] = (bf16_t)src[j];
    }
    v16bf bb0 = load_b_frag(brow0, kb, hi);
    v16bf bb1 = load_b_frag(brow1, kb, hi);
    acc0 = WMMA_BF16(a, bb0, acc0);
    acc1 = WMMA_BF16(a, bb1, acc1);
  }
  float bia0 = b1[l15], bia1 = b1[16 + l15];
  int mm = mt * 16 + (hi ? 8 : 0);
  int opy = mm / 20, opx = mm - opy * 20;     // incremental, no per-r division
#pragma unroll
  for (int r = 0; r < 8; ++r) {
    size_t o = (size_t)sample * 12800 + opy * 20 + opx;
    float v0 = acc0[r] + bia0; v0 = v0 > 0.f ? v0 : 0.f;
    float v1 = acc1[r] + bia1; v1 = v1 > 0.f ? v1 : 0.f;
    out[o + (size_t)l15 * 400]        = (bf16_t)v0;
    out[o + (size_t)(16 + l15) * 400] = (bf16_t)v1;
    if (++opx == 20) { opx = 0; ++opy; }
  }
}

// ---------------------------------------------------------------------------
// Conv2: one workgroup (192 thr = 6 waves) per sample.  Whole per-sample input
// [32][20][20] bf16 (25.6 KB) DMA'd to LDS by the Tensor Data Mover (wave 0
// issues tensor_load_to_lds, waits TENSORcnt, then workgroup barrier).
// out [64][9][9] bf16, 4x4 stride 2.  K = c*16+ky*4+kx (512).
// ---------------------------------------------------------------------------
__global__ void conv2_wmma(const bf16_t* __restrict__ in, const bf16_t* __restrict__ wb2,
                           const float* __restrict__ b2, bf16_t* __restrict__ out) {
  __shared__ bf16_t tile[12800];
  int sample = blockIdx.x;
  const bf16_t* sin = in + (size_t)sample * 12800;

  if ((threadIdx.x >> 5) == 0) {
    // D# group 0: count=1, lds_addr=0, global_addr=sin, type=2 ("image")
    uint64_t ga = (uint64_t)(uintptr_t)sin;
    u32x4 g0;
    g0[0] = 1u;                                     // count=1, user mode
    g0[1] = 0u;                                     // lds_addr = 0 (tile base)
    g0[2] = (uint32_t)ga;                           // global_addr[31:0]
    g0[3] = (uint32_t)((ga >> 32) & 0x01FFFFFFu) | (2u << 30);  // [56:32] | type=2
    // D# group 1: data_size=2B, 1 row of 12800 elements
    u32x8 g1;
    g1[0] = (1u << 16);                             // data_size=1 (2 bytes)
    g1[1] = (12800u & 0xFFFFu) << 16;               // tensor_dim0[15:0] @ bits 63:48
    g1[2] = (12800u >> 16) | (1u << 16);            // tensor_dim0[31:16], tensor_dim1=1
    g1[3] = (12800u << 16);                         // tensor_dim1 hi=0, tile_dim0=12800
    g1[4] = 1u;                                     // tile_dim1=1, tile_dim2=0
    g1[5] = 12800u;                                 // tensor_dim0_stride[31:0]
    g1[6] = 0u;                                     // stride hi, tensor_dim1_stride lo
    g1[7] = 0u;
    asm volatile("tensor_load_to_lds %0, %1" :: "s"(g0), "s"(g1) : "memory");
    __builtin_amdgcn_s_wait_tensorcnt(0);
  }
  __syncthreads();

  int mt = threadIdx.x >> 5;                   // 6 waves -> m-tiles 0..5
  int lane = threadIdx.x & 31, l15 = lane & 15, hi = lane >> 4;
  int m = mt * 16 + l15;
  int mq = (m < 81) ? m : 0;
  int py = mq / 9, px = mq - py * 9;
  v8f acc[4];
#pragma unroll
  for (int nt = 0; nt < 4; ++nt) acc[nt] = (v8f){};
  for (int ks = 0; ks < 16; ++ks) {
    int kb = ks * 32;
    v16bf a;
#pragma unroll
    for (int c = 0; c < 2; ++c) {
      int kc = kb + (hi ? 8 : 0) + c * 16;     // 8 K: fixed ch, rows ky0,ky0+1, kx 0..3
      int ch = kc >> 4, ky0 = (kc >> 2) & 3;
      const bf16_t* p0 = tile + ch * 400 + (py * 2 + ky0) * 20 + px * 2;
#pragma unroll
      for (int j = 0; j < 4; ++j) { a[c * 8 + j] = p0[j]; a[c * 8 + 4 + j] = p0[20 + j]; }
    }
#pragma unroll
    for (int nt = 0; nt < 4; ++nt) {
      v16bf bb = load_b_frag(wb2 + (size_t)(nt * 16 + l15) * 512, kb, hi);
      acc[nt] = WMMA_BF16(a, bb, acc[nt]);
    }
  }
  int mm = mt * 16 + (hi ? 8 : 0);
  int opy = mm / 9, opx = mm - opy * 9;
#pragma unroll
  for (int r = 0; r < 8; ++r) {
    if (mm + r < 81) {
#pragma unroll
      for (int nt = 0; nt < 4; ++nt) {
        int ch = nt * 16 + l15;
        float v = acc[nt][r] + b2[ch]; v = v > 0.f ? v : 0.f;
        out[(size_t)sample * 5184 + ch * 81 + opy * 9 + opx] = (bf16_t)v;
      }
    }
    if (++opx == 9) { opx = 0; ++opy; }
  }
}

// ---------------------------------------------------------------------------
// Conv3: one workgroup (128 thr = 4 waves) per sample.  Input [64][9][9] bf16
// (10.4 KB) staged to LDS with async global->LDS b128 (ASYNCcnt).
// out flat [3136] bf16 (c*49+py*7+px), 3x3 s1.  K = c*9+ky*3+kx (576).
// ---------------------------------------------------------------------------
__global__ void conv3_wmma(const bf16_t* __restrict__ in, const bf16_t* __restrict__ wb3,
                           const float* __restrict__ b3, bf16_t* __restrict__ out) {
  __shared__ bf16_t tile[5184];
  int sample = blockIdx.x;
  const bf16_t* sin = in + (size_t)sample * 5184;
  // 648 x 16-byte chunks, one async lane-load each; LDS dest = lane VGPR offset
  for (int i = threadIdx.x; i < 648; i += blockDim.x) {
    int ldsoff = i * 16;
    const char* g = (const char*)sin + (size_t)i * 16;
    asm volatile("global_load_async_to_lds_b128 %0, %1, off"
                 :: "v"(ldsoff), "v"(g) : "memory");
  }
  asm volatile("s_wait_asynccnt 0x0" ::: "memory");
  __syncthreads();

  int mt = threadIdx.x >> 5;                   // 4 waves -> m-tiles 0..3
  int lane = threadIdx.x & 31, l15 = lane & 15, hi = lane >> 4;
  int m = mt * 16 + l15;
  int mq = (m < 49) ? m : 0;
  int py = mq / 7, px = mq - py * 7;
  v8f acc[4];
#pragma unroll
  for (int nt = 0; nt < 4; ++nt) acc[nt] = (v8f){};
  for (int ks = 0; ks < 18; ++ks) {
    int kb = ks * 32;
    v16bf a;
#pragma unroll
    for (int j = 0; j < 16; ++j) {
      int k = kb + (hi ? 8 : 0) + j + ((j >= 8) ? 8 : 0);   // Kof(j)
      int c = k / 9, r9 = k - c * 9;
      int ky = r9 / 3, kx = r9 - ky * 3;
      a[j] = tile[c * 81 + (py + ky) * 9 + (px + kx)];
    }
#pragma unroll
    for (int nt = 0; nt < 4; ++nt) {
      v16bf bb = load_b_frag(wb3 + (size_t)(nt * 16 + l15) * 576, kb, hi);
      acc[nt] = WMMA_BF16(a, bb, acc[nt]);
    }
  }
  int mm = mt * 16 + (hi ? 8 : 0);
  int opy = mm / 7, opx = mm - opy * 7;
#pragma unroll
  for (int r = 0; r < 8; ++r) {
    if (mm + r < 49) {
#pragma unroll
      for (int nt = 0; nt < 4; ++nt) {
        int ch = nt * 16 + l15;
        float v = acc[nt][r] + b3[ch]; v = v > 0.f ? v : 0.f;
        out[(size_t)sample * 3136 + ch * 49 + opy * 7 + opx] = (bf16_t)v;
      }
    }
    if (++opx == 7) { opx = 0; ++opy; }
  }
}

// --------------------------- small utility kernels -------------------------

__global__ void cvt_f32_bf16(const float* __restrict__ src, bf16_t* __restrict__ dst, int n) {
  int i = blockIdx.x * blockDim.x + threadIdx.x;
  if (i < n) dst[i] = (bf16_t)src[i];
}

// W_ih [1536][531] f32 -> [1536][544] bf16 zero-padded
__global__ void pad_wih(const float* __restrict__ src, bf16_t* __restrict__ dst) {
  int i = blockIdx.x * blockDim.x + threadIdx.x;
  if (i >= 1536 * 544) return;
  int r = i / 544, c = i - r * 544;
  dst[i] = (c < 531) ? (bf16_t)src[r * 531 + c] : (bf16_t)0.f;
}

// xin cols 512..543: one-hot(action)*mask (18), reward*mask (1), zeros (pad)
__global__ void xin_misc(const int* __restrict__ action, const float* __restrict__ reward,
                         const int* __restrict__ done, bf16_t* __restrict__ xin) {
  int i = blockIdx.x * blockDim.x + threadIdx.x;
  if (i >= 4096 * 32) return;
  int n = i >> 5, j = i & 31;
  int t = n >> 6, b = n & 63;
  int src = (t + 3) * 64 + b;
  float mf = (float)(1 - done[src]);
  float v = 0.f;
  if (j < 18)       v = (action[src] == j) ? mf : 0.f;
  else if (j == 18) v = reward[src] * mf;
  xin[(size_t)n * 544 + 512 + j] = (bf16_t)v;
}

// h_masked (bf16) for the recurrent GEMM of step t
__global__ void gru_mask(const float* __restrict__ h, const int* __restrict__ done,
                         bf16_t* __restrict__ hm, int t) {
  int i = blockIdx.x * blockDim.x + threadIdx.x;
  if (i >= 64 * 512) return;
  int b = i >> 9;
  float mf = (float)(1 - done[(t + 3) * 64 + b]);
  hm[i] = (bf16_t)(h[i] * mf);
}

// GRU gate math for step t; h updated in place, y emitted as bf16
__global__ void gru_gate(const float* __restrict__ gx, const float* __restrict__ gh,
                         const int* __restrict__ done, float* __restrict__ h,
                         bf16_t* __restrict__ ybf, int t) {
  int i = blockIdx.x * blockDim.x + threadIdx.x;
  if (i >= 64 * 512) return;
  int b = i >> 9, j = i & 511;
  float mf = (float)(1 - done[(t + 3) * 64 + b]);
  float hm = h[i] * mf;
  const float* gxr = gx + (size_t)(t * 64 + b) * 1536;
  const float* ghr = gh + (size_t)b * 1536;
  float xr = gxr[j], xz = gxr[j + 512], xn = gxr[j + 1024];
  float hr = ghr[j], hz = ghr[j + 512], hnv = ghr[j + 1024];
  float rg = 1.f / (1.f + __expf(-(xr + hr)));
  float zg = 1.f / (1.f + __expf(-(xz + hz)));
  float ng = tanhf(xn + rg * hnv);
  float hn2 = (1.f - zg) * ng + zg * hm;
  h[i] = hn2;
  ybf[(size_t)(t * 64 + b) * 512 + j] = (bf16_t)hn2;
}

// Final projections, one wave per output row m: lanes split K, reduce with
// __shfl_xor (LDS-crossbar permutes on wave32).
__global__ void head_proj_wave(const bf16_t* __restrict__ advh, const bf16_t* __restrict__ valh,
                               const float* __restrict__ adv2_w, const float* __restrict__ val2_w,
                               const float* __restrict__ val2_b,
                               float* __restrict__ adv_raw, float* __restrict__ val_raw) {
  int m = blockIdx.x * (blockDim.x >> 5) + (threadIdx.x >> 5);   // 4096 waves
  int lane = threadIdx.x & 31;
  float av[16], vv[16];
  const bf16_t* arow = advh + (size_t)m * 512 + lane * 16;
  const bf16_t* vrow = valh + (size_t)m * 512 + lane * 16;
#pragma unroll
  for (int j = 0; j < 16; ++j) { av[j] = (float)arow[j]; vv[j] = (float)vrow[j]; }
  for (int a = 0; a < 18; ++a) {
    const float* w = adv2_w + (size_t)a * 512 + lane * 16;
    float s = 0.f;
#pragma unroll
    for (int j = 0; j < 16; ++j) s += av[j] * w[j];
#pragma unroll
    for (int o = 16; o > 0; o >>= 1) s += __shfl_xor(s, o, 32);
    if (lane == 0) adv_raw[(size_t)m * 18 + a] = s;
  }
  const float* w = val2_w + lane * 16;
  float s = 0.f;
#pragma unroll
  for (int j = 0; j < 16; ++j) s += vv[j] * w[j];
#pragma unroll
  for (int o = 16; o > 0; o >>= 1) s += __shfl_xor(s, o, 32);
  if (lane == 0) val_raw[m] = s + val2_b[0];
}

// q = val + adv - mean_a(adv)
__global__ void head_combine(const float* __restrict__ adv_raw, const float* __restrict__ val_raw,
                             float* __restrict__ q) {
  int i = blockIdx.x * blockDim.x + threadIdx.x;
  if (i >= 4096 * 18) return;
  int m = i / 18, a = i - m * 18;
  float mean = 0.f;
  const float* row = adv_raw + (size_t)m * 18;
#pragma unroll
  for (int k = 0; k < 18; ++k) mean += row[k];
  mean *= (1.f / 18.f);
  q[i] = val_raw[m] + row[a] - mean;
}

__global__ void copy_f32(const float* __restrict__ src, float* __restrict__ dst, int n) {
  int i = blockIdx.x * blockDim.x + threadIdx.x;
  if (i < n) dst[i] = src[i];
}

// ---------------------------------------------------------------------------

extern "C" void kernel_launch(void* const* d_in, const int* in_sizes, int n_in,
                              void* d_out, int out_size, void* d_ws, size_t ws_size,
                              hipStream_t stream) {
  (void)in_sizes; (void)n_in; (void)out_size; (void)ws_size;
  const float* obs    = (const float*)d_in[0];
  const int*   action = (const int*)  d_in[1];
  const float* reward = (const float*)d_in[2];
  const int*   done   = (const int*)  d_in[3];
  const float* hx     = (const float*)d_in[4];
  const float* w1     = (const float*)d_in[5];
  const float* b1     = (const float*)d_in[6];
  const float* w2     = (const float*)d_in[7];
  const float* b2     = (const float*)d_in[8];
  const float* w3     = (const float*)d_in[9];
  const float* b3     = (const float*)d_in[10];
  const float* fc_w   = (const float*)d_in[11];
  const float* fc_b   = (const float*)d_in[12];
  const float* W_ih   = (const float*)d_in[13];
  const float* W_hh   = (const float*)d_in[14];
  const float* b_ih   = (const float*)d_in[15];
  const float* b_hh   = (const float*)d_in[16];
  const float* adv1_w = (const float*)d_in[17];
  const float* adv1_b = (const float*)d_in[18];
  const float* adv2_w = (const float*)d_in[19];
  const float* val1_w = (const float*)d_in[20];
  const float* val1_b = (const float*)d_in[21];
  const float* val2_w = (const float*)d_in[22];
  const float* val2_b = (const float*)d_in[23];

  // workspace carve-up (256B aligned)
  size_t off = 0;
  char* base = (char*)d_ws;
  auto carve = [&](size_t bytes) -> void* {
    off = (off + 255) & ~(size_t)255;
    void* p = base + off; off += bytes; return p;
  };
  bf16_t* wb1   = (bf16_t*)carve(32   * 256  * 2);
  bf16_t* wb2   = (bf16_t*)carve(64   * 512  * 2);
  bf16_t* wb3   = (bf16_t*)carve(64   * 576  * 2);
  bf16_t* fcwb  = (bf16_t*)carve(512  * 3136 * 2);
  bf16_t* wihb  = (bf16_t*)carve(1536 * 544  * 2);
  bf16_t* whhb  = (bf16_t*)carve(1536 * 512  * 2);
  bf16_t* adv1b = (bf16_t*)carve(512  * 512  * 2);
  bf16_t* val1b = (bf16_t*)carve(512  * 512  * 2);
  bf16_t* c1    = (bf16_t*)carve((size_t)4096 * 12800 * 2);
  bf16_t* c2    = (bf16_t*)carve((size_t)4096 * 5184  * 2);
  bf16_t* c3    = (bf16_t*)carve((size_t)4096 * 3136  * 2);
  bf16_t* xin   = (bf16_t*)carve((size_t)4096 * 544   * 2);
  float*  gx    = (float*) carve((size_t)4096 * 1536  * 4);
  float*  hbuf  = (float*) carve(64 * 512 * 4);
  bf16_t* hmb   = (bf16_t*)carve(64 * 512 * 2);
  float*  gh    = (float*) carve(64 * 1536 * 4);
  bf16_t* ybf   = (bf16_t*)carve((size_t)4096 * 512 * 2);
  bf16_t* advh  = (bf16_t*)carve((size_t)4096 * 512 * 2);
  bf16_t* valh  = (bf16_t*)carve((size_t)4096 * 512 * 2);
  float*  araw  = (float*) carve((size_t)4096 * 18 * 4);
  float*  vraw  = (float*) carve((size_t)4096 * 4);

  float* q_out  = (float*)d_out;             // [4096][18]
  float* h_out  = (float*)d_out + 4096 * 18; // [64][512]

  const int TB = 256;
  auto blk = [](int n, int tb) { return (n + tb - 1) / tb; };

  // 1) weights -> bf16
  cvt_f32_bf16<<<blk(8192,   TB), TB, 0, stream>>>(w1,     wb1,   8192);
  cvt_f32_bf16<<<blk(32768,  TB), TB, 0, stream>>>(w2,     wb2,   32768);
  cvt_f32_bf16<<<blk(36864,  TB), TB, 0, stream>>>(w3,     wb3,   36864);
  cvt_f32_bf16<<<blk(1605632,TB), TB, 0, stream>>>(fc_w,   fcwb,  1605632);
  cvt_f32_bf16<<<blk(786432, TB), TB, 0, stream>>>(W_hh,   whhb,  786432);
  cvt_f32_bf16<<<blk(262144, TB), TB, 0, stream>>>(adv1_w, adv1b, 262144);
  cvt_f32_bf16<<<blk(262144, TB), TB, 0, stream>>>(val1_w, val1b, 262144);
  pad_wih<<<blk(1536 * 544, TB), TB, 0, stream>>>(W_ih, wihb);

  // 2) conv stack (implicit GEMM, WMMA); conv2 stages via TDM, conv3 via
  //    async global->LDS
  conv1_wmma<<<4096 * 25 / 8, TB, 0, stream>>>(obs, wb1, b1, c1);
  conv2_wmma<<<4096, 192, 0, stream>>>(c1, wb2, b2, c2);
  conv3_wmma<<<4096, 128, 0, stream>>>(c2, wb3, b3, c3);

  // 3) FC 3136->512 (relu) writes xin[:, 0:512] as bf16.  <2,4>: 1024 waves.
  gemm_bf16_wmma<2, 4><<<blk(1024, 8), TB, 0, stream>>>(
      c3, fcwb, fc_b, nullptr, xin, 3136, 98, 8, 1024, 544, 1);
  xin_misc<<<blk(4096 * 32, TB), TB, 0, stream>>>(action, reward, done, xin);

  // 4) gx = xin @ W_ih^T + b_ih for all steps.  mgrp=128, ngrp=24 -> 3072.
  gemm_bf16_wmma<2, 4><<<blk(3072, 8), TB, 0, stream>>>(
      xin, wihb, b_ih, gx, nullptr, 544, 17, 24, 3072, 1536, 0);

  // 5) GRU scan (sequential over 64 steps); recurrent GEMM is M=64 -> <4,4>.
  hipMemcpyAsync(hbuf, hx, 64 * 512 * sizeof(float), hipMemcpyDeviceToDevice, stream);
  for (int t = 0; t < 64; ++t) {
    gru_mask<<<blk(64 * 512, TB), TB, 0, stream>>>(hbuf, done, hmb, t);
    gemm_bf16_wmma<4, 4><<<blk(24, 8), TB, 0, stream>>>(
        hmb, whhb, b_hh, gh, nullptr, 512, 16, 24, 24, 1536, 0);
    gru_gate<<<blk(64 * 512, TB), TB, 0, stream>>>(gx, gh, done, hbuf, ybf, t);
  }

  // 6) dueling heads: 512x512 relu layers via WMMA (<2,4>: 1024 waves each)
  gemm_bf16_wmma<2, 4><<<blk(1024, 8), TB, 0, stream>>>(
      ybf, adv1b, adv1_b, nullptr, advh, 512, 16, 8, 1024, 512, 1);
  gemm_bf16_wmma<2, 4><<<blk(1024, 8), TB, 0, stream>>>(
      ybf, val1b, val1_b, nullptr, valh, 512, 16, 8, 1024, 512, 1);

  // 7) final projections (wave-per-row, shuffle reduce) + dueling combine
  head_proj_wave<<<blk(4096 * 32, TB), TB, 0, stream>>>(
      advh, valh, adv2_w, val2_w, val2_b, araw, vraw);
  head_combine<<<blk(4096 * 18, TB), TB, 0, stream>>>(araw, vraw, q_out);
  copy_f32<<<blk(64 * 512, TB), TB, 0, stream>>>(hbuf, h_out, 64 * 512);
}
